// MHA_81260781240496
// MI455X (gfx1250) — matrix operand order repaired and checked by reading, not compiled, not run on hardware
//
#include <hip/hip_runtime.h>
#include <hip/hip_bf16.h>

#define S_LEN 4096
#define EMB   1024
#define NH    16
#define HD    64

typedef __attribute__((ext_vector_type(16))) __bf16   v16bf;
typedef __attribute__((ext_vector_type(8)))  float    v8f;
typedef __attribute__((ext_vector_type(4)))  unsigned v4u;
typedef __attribute__((ext_vector_type(8)))  int      v8i;
typedef __attribute__((ext_vector_type(4)))  int      v4i;

union FragAB { v16bf v; unsigned u[8]; };
union FragC  { v8f   v; float    f[8]; };

#if defined(__has_builtin)
#  if __has_builtin(__builtin_amdgcn_tensor_load_to_lds)
#    define HAVE_TDM 1
#  endif
#endif
#ifndef HAVE_TDM
#  define HAVE_TDM 0
#endif

__device__ __forceinline__ unsigned short f2bf(float x) {
  unsigned u = __float_as_uint(x);
  u += 0x7FFFu + ((u >> 16) & 1u);   // round-to-nearest-even
  return (unsigned short)(u >> 16);
}

// A-matrix (16x32 bf16) K-pair offset per ISA 7.12.2 table
__device__ __forceinline__ int kOffA(int v, int hi) {
  return ((v & 4) << 2) + ((v & 3) << 1) + (hi ? 8 : 0);
}
// B-matrix (32x16 bf16) K-pair offset (same K order as A)
__device__ __forceinline__ int kOffB(int v, int hi) {
  return ((v & 4) << 2) + ((v & 3) << 1) + (hi ? 8 : 0);
}

__device__ __forceinline__ v16bf loadFragA(const unsigned short* row, int hi) {
  FragAB f;
#pragma unroll
  for (int v = 0; v < 8; ++v)
    f.u[v] = *(const unsigned*)(row + kOffA(v, hi));
  return f.v;
}
__device__ __forceinline__ v16bf loadFragB(const unsigned short* row, int hi) {
  FragAB f;
#pragma unroll
  for (int v = 0; v < 8; ++v)
    f.u[v] = *(const unsigned*)(row + kOffB(v, hi));
  return f.v;
}

__device__ __forceinline__ v8f wmma_bf16(v16bf a, v16bf b, v8f c) {
  return __builtin_amdgcn_wmma_f32_16x16x32_bf16(false, a, false, b, (short)0, c,
                                                 false, false);
}

#if HAVE_TDM
__device__ __forceinline__ unsigned lds_offset(const void* p) {
  return (unsigned)(size_t)(__attribute__((address_space(3))) const void*)p;
}

// TDM: load a 2D tile of bf16 (tile_d1 rows x tile_d0 contiguous elems,
// row stride stride_d0 elems) from global into LDS, packed row-major.
// D# layout per CDNA5 ISA 8.3/8.4: group0 {count=1, lds_addr, global_addr,
// type=2}; group1 {data_size=2B, tensor dims, tile dims, dim0 stride}.
__device__ __forceinline__ void tdm_load_2d_bf16(
    const unsigned short* gsrc, const void* ldst,
    unsigned tile_d0, unsigned tile_d1,
    unsigned tensor_d0, unsigned tensor_d1, unsigned stride_d0) {
  unsigned long long ga = (unsigned long long)gsrc;
  v4u g0;
  g0[0] = 1u;                                       // count=1, user desc
  g0[1] = lds_offset(ldst);                         // lds_addr (bytes)
  g0[2] = (unsigned)ga;                             // global_addr[31:0]
  g0[3] = (unsigned)((ga >> 32) & 0x1FFFFFFu) | (2u << 30);  // addr[56:32]|type=2
  v8i g1;
  g1[0] = 0x00010000;                               // data_size=1 (2 bytes)
  g1[1] = (int)((tensor_d0 & 0xFFFFu) << 16);       // tensor_dim0[15:0]
  g1[2] = (int)((tensor_d0 >> 16) | ((tensor_d1 & 0xFFFFu) << 16));
  g1[3] = (int)((tensor_d1 >> 16) | (tile_d0 << 16));
  g1[4] = (int)(tile_d1 & 0xFFFFu);                 // tile_dim1; tile_dim2=0
  g1[5] = (int)stride_d0;                           // tensor_dim0_stride lo32
  g1[6] = 0;
  g1[7] = 0;
  v4i z4 = {};
#if __clang_major__ >= 23
  v8i z8 = {};
  __builtin_amdgcn_tensor_load_to_lds(g0, g1, z4, z4, z8, 0);
#else
  __builtin_amdgcn_tensor_load_to_lds(g0, g1, z4, z4, 0);
#endif
}
#endif  // HAVE_TDM

// ---------------------------------------------------------------- converts
__global__ void convert_f32_bf16(const float* __restrict__ src,
                                 unsigned short* __restrict__ dst, int n) {
  int i = blockIdx.x * blockDim.x + threadIdx.x;
  int stride = gridDim.x * blockDim.x;
  for (; i < n; i += stride) dst[i] = f2bf(src[i]);
}

// ---------------------------------------------------------------- QKV GEMM
// x(4096x1024) @ Wqkv(1024x3072) + bqkv -> Q,K,V bf16 [head][s][d], Q prescaled
__global__ __launch_bounds__(256) void qkv_gemm(
    const float* __restrict__ x, const unsigned short* __restrict__ Wbf,
    const float* __restrict__ bqkv,
    unsigned short* __restrict__ qws, unsigned short* __restrict__ kws,
    unsigned short* __restrict__ vws) {
  __shared__ __align__(16) unsigned short As[128 * 64];  // 16 KB
  __shared__ __align__(16) unsigned short Bt[64 * 64];   //  8 KB
  const int tid = threadIdx.x;
  const int lane = tid & 31, wid = tid >> 5;
  const int hi = (lane >> 4) & 1, l15 = lane & 15;
  const int bM = blockIdx.y * 128, bN = blockIdx.x * 64;
  const int wm = (wid >> 1) * 32, wn = (wid & 1) * 32;

  FragC acc[2][2];
#pragma unroll
  for (int mi = 0; mi < 2; ++mi)
#pragma unroll
    for (int ni = 0; ni < 2; ++ni)
#pragma unroll
      for (int r = 0; r < 8; ++r) acc[mi][ni].f[r] = 0.f;

  for (int kk = 0; kk < EMB; kk += 64) {
    if (kk + 64 < EMB)
      __builtin_prefetch(&x[(size_t)(bM + (tid >> 1)) * EMB + kk + 64], 0, 0);
#pragma unroll
    for (int i = 0; i < 32; ++i) {                  // 8192 fp32 -> bf16
      int idx = tid + i * 256;
      int row = idx >> 6, col = idx & 63;
      As[idx] = f2bf(x[(size_t)(bM + row) * EMB + kk + col]);
    }
#pragma unroll
    for (int i = 0; i < 16; ++i) {                  // B transposed into [n][k]
      int idx = tid + i * 256;
      int k = idx >> 6, n = idx & 63;
      Bt[n * 64 + k] = Wbf[(size_t)(kk + k) * (3 * EMB) + bN + n];
    }
    __syncthreads();
    v16bf a00 = loadFragA(&As[(wm + l15) * 64 + 0], hi);
    v16bf a01 = loadFragA(&As[(wm + l15) * 64 + 32], hi);
    v16bf a10 = loadFragA(&As[(wm + 16 + l15) * 64 + 0], hi);
    v16bf a11 = loadFragA(&As[(wm + 16 + l15) * 64 + 32], hi);
    v16bf b00 = loadFragB(&Bt[(wn + l15) * 64 + 0], hi);
    v16bf b01 = loadFragB(&Bt[(wn + l15) * 64 + 32], hi);
    v16bf b10 = loadFragB(&Bt[(wn + 16 + l15) * 64 + 0], hi);
    v16bf b11 = loadFragB(&Bt[(wn + 16 + l15) * 64 + 32], hi);
    acc[0][0].v = wmma_bf16(a00, b00, acc[0][0].v);
    acc[0][0].v = wmma_bf16(a01, b01, acc[0][0].v);
    acc[0][1].v = wmma_bf16(a00, b10, acc[0][1].v);
    acc[0][1].v = wmma_bf16(a01, b11, acc[0][1].v);
    acc[1][0].v = wmma_bf16(a10, b00, acc[1][0].v);
    acc[1][0].v = wmma_bf16(a11, b01, acc[1][0].v);
    acc[1][1].v = wmma_bf16(a10, b10, acc[1][1].v);
    acc[1][1].v = wmma_bf16(a11, b11, acc[1][1].v);
    __syncthreads();
  }

#pragma unroll
  for (int mi = 0; mi < 2; ++mi)
#pragma unroll
    for (int ni = 0; ni < 2; ++ni)
#pragma unroll
      for (int r = 0; r < 8; ++r) {
        int m = bM + wm + mi * 16 + (hi ? 8 : 0) + r;
        int n = bN + wn + ni * 16 + l15;
        float val = acc[mi][ni].f[r] + bqkv[n];
        int which = n >> 10;
        int e = n & 1023;
        int h = e >> 6, d = e & 63;
        size_t off = ((size_t)h * S_LEN + m) * HD + d;
        unsigned short* base = (which == 0) ? qws : ((which == 1) ? kws : vws);
        float sc = (which == 0) ? 0.125f : 1.0f;    // fold 1/sqrt(HD) into Q
        base[off] = f2bf(val * sc);
      }
}

// ---------------------------------------------------------------- flash attn
// 1 block = 64 query rows of one head; 4 waves x 16 rows; causal online softmax
__global__ __launch_bounds__(128) void flash_attn(
    const unsigned short* __restrict__ qws, const unsigned short* __restrict__ kws,
    const unsigned short* __restrict__ vws, unsigned short* __restrict__ ctx) {
  __shared__ __align__(16) unsigned short Kl[64 * 64];      // [key][d]
  __shared__ __align__(16) unsigned short Vt[64 * 64];      // [d][key]
  __shared__ __align__(16) unsigned short Pl[4][16 * 64];   // per-wave P
  const int tid = threadIdx.x;
  const int lane = tid & 31, wid = tid >> 5;
  const int hi = (lane >> 4) & 1, l15 = lane & 15;
  const int h = blockIdx.y;
  const int qBase = blockIdx.x * 64;
  const unsigned short* qh = qws + (size_t)h * S_LEN * HD;
  const unsigned short* kh = kws + (size_t)h * S_LEN * HD;
  const unsigned short* vh = vws + (size_t)h * S_LEN * HD;

  v16bf q0 = loadFragA(&qh[(size_t)(qBase + wid * 16 + l15) * HD + 0], hi);
  v16bf q1 = loadFragA(&qh[(size_t)(qBase + wid * 16 + l15) * HD + 32], hi);

  FragC o[4];
  float mrow[8], lrow[8];
#pragma unroll
  for (int dt = 0; dt < 4; ++dt)
#pragma unroll
    for (int r = 0; r < 8; ++r) o[dt].f[r] = 0.f;
#pragma unroll
  for (int r = 0; r < 8; ++r) { mrow[r] = -1e30f; lrow[r] = 0.f; }

  const int nKB = (qBase >> 6) + 1;  // causal: only key blocks <= query block
  for (int j = 0; j < nKB; ++j) {
    const int kBase = j * 64;
#if HAVE_TDM
    if (tid < 32) {  // wave 0 drives the Tensor Data Mover for the K tile
      tdm_load_2d_bf16(&kh[(size_t)kBase * HD], Kl, 64, 64, 64, 64, 64);
      __builtin_amdgcn_s_wait_tensorcnt(0);
    }
#else
    {  // K tile straight copy (8 KB as dwords)
      const unsigned* src = (const unsigned*)&kh[(size_t)kBase * HD];
      unsigned* dst = (unsigned*)Kl;
#pragma unroll
      for (int i = 0; i < 16; ++i) dst[tid + i * 128] = src[tid + i * 128];
    }
#endif
#pragma unroll
    for (int i = 0; i < 32; ++i) {  // V transposed (TDM cannot transpose)
      int idx = tid + i * 128;
      int key = idx >> 6, d = idx & 63;
      Vt[d * 64 + key] = vh[(size_t)(kBase + key) * HD + d];
    }
    __syncthreads();

    // S = Q(16x64) @ K^T  -> 4 subtiles of 16x16
    FragC c[4];
#pragma unroll
    for (int nt = 0; nt < 4; ++nt) {
#pragma unroll
      for (int r = 0; r < 8; ++r) c[nt].f[r] = 0.f;
      v16bf kb0 = loadFragB(&Kl[(nt * 16 + l15) * 64 + 0], hi);
      v16bf kb1 = loadFragB(&Kl[(nt * 16 + l15) * 64 + 32], hi);
      c[nt].v = wmma_bf16(q0, kb0, c[nt].v);
      c[nt].v = wmma_bf16(q1, kb1, c[nt].v);
    }

    // online softmax (rows live across 16 lanes of a half-wave)
    const int qRow0 = qBase + wid * 16 + (hi ? 8 : 0);
#pragma unroll
    for (int r = 0; r < 8; ++r) {
      const int qrow = qRow0 + r;
      float rm = -1e30f;
#pragma unroll
      for (int nt = 0; nt < 4; ++nt) {
        float s = c[nt].f[r];
        if (kBase + nt * 16 + l15 > qrow) s -= 10000.f;  // causal mask
        c[nt].f[r] = s;
        rm = fmaxf(rm, s);
      }
#pragma unroll
      for (int off = 1; off < 16; off <<= 1) rm = fmaxf(rm, __shfl_xor(rm, off, 16));
      float mnew = fmaxf(mrow[r], rm);
      float corr = __expf(mrow[r] - mnew);
      float rsum = 0.f;
#pragma unroll
      for (int nt = 0; nt < 4; ++nt) {
        float p = __expf(c[nt].f[r] - mnew);
        c[nt].f[r] = p;
        rsum += p;
      }
#pragma unroll
      for (int off = 1; off < 16; off <<= 1) rsum += __shfl_xor(rsum, off, 16);
      lrow[r] = lrow[r] * corr + rsum;
      mrow[r] = mnew;
#pragma unroll
      for (int dt = 0; dt < 4; ++dt) o[dt].f[r] *= corr;
    }

    // P (C-layout) -> per-wave LDS -> A-layout fragments
#pragma unroll
    for (int nt = 0; nt < 4; ++nt)
#pragma unroll
      for (int r = 0; r < 8; ++r) {
        int m = (hi ? 8 : 0) + r;
        Pl[wid][m * 64 + nt * 16 + l15] = f2bf(c[nt].f[r]);
      }

    v16bf p0 = loadFragA(&Pl[wid][l15 * 64 + 0], hi);
    v16bf p1 = loadFragA(&Pl[wid][l15 * 64 + 32], hi);
#pragma unroll
    for (int dt = 0; dt < 4; ++dt) {
      v16bf vb0 = loadFragB(&Vt[(dt * 16 + l15) * 64 + 0], hi);
      v16bf vb1 = loadFragB(&Vt[(dt * 16 + l15) * 64 + 32], hi);
      o[dt].v = wmma_bf16(p0, vb0, o[dt].v);
      o[dt].v = wmma_bf16(p1, vb1, o[dt].v);
    }
    __syncthreads();
  }

  // normalize and store context in [s][E] bf16
#pragma unroll
  for (int dt = 0; dt < 4; ++dt)
#pragma unroll
    for (int r = 0; r < 8; ++r) {
      int m = qBase + wid * 16 + (hi ? 8 : 0) + r;
      float val = o[dt].f[r] / lrow[r];
      ctx[(size_t)m * EMB + h * HD + dt * 16 + l15] = f2bf(val);
    }
}

// ---------------------------------------------------------------- out GEMM
__global__ __launch_bounds__(256) void out_gemm(
    const unsigned short* __restrict__ ctx, const unsigned short* __restrict__ Wbf,
    const float* __restrict__ bout, float* __restrict__ out) {
  __shared__ __align__(16) unsigned short As[128 * 64];  // 16 KB
  __shared__ __align__(16) unsigned short Bt[64 * 64];   //  8 KB
  const int tid = threadIdx.x;
  const int lane = tid & 31, wid = tid >> 5;
  const int hi = (lane >> 4) & 1, l15 = lane & 15;
  const int bM = blockIdx.y * 128, bN = blockIdx.x * 64;
  const int wm = (wid >> 1) * 32, wn = (wid & 1) * 32;

  FragC acc[2][2];
#pragma unroll
  for (int mi = 0; mi < 2; ++mi)
#pragma unroll
    for (int ni = 0; ni < 2; ++ni)
#pragma unroll
      for (int r = 0; r < 8; ++r) acc[mi][ni].f[r] = 0.f;

  for (int kk = 0; kk < EMB; kk += 64) {
#if HAVE_TDM
    if (tid < 32) {  // wave 0 drives the TDM for the A tile (128x64 bf16)
      tdm_load_2d_bf16(&ctx[(size_t)bM * EMB + kk], As, 64, 128, 64, 128, EMB);
      __builtin_amdgcn_s_wait_tensorcnt(0);
    }
#else
#pragma unroll
    for (int i = 0; i < 16; ++i) {  // A: bf16 dword copy, 4096 dwords
      int idx = tid + i * 256;
      int row = idx >> 5, c2 = idx & 31;
      ((unsigned*)&As[row * 64])[c2] =
          ((const unsigned*)&ctx[(size_t)(bM + row) * EMB + kk])[c2];
    }
#endif
#pragma unroll
    for (int i = 0; i < 16; ++i) {
      int idx = tid + i * 256;
      int k = idx >> 6, n = idx & 63;
      Bt[n * 64 + k] = Wbf[(size_t)(kk + k) * EMB + bN + n];
    }
    __syncthreads();
    v16bf a00 = loadFragA(&As[(wm + l15) * 64 + 0], hi);
    v16bf a01 = loadFragA(&As[(wm + l15) * 64 + 32], hi);
    v16bf a10 = loadFragA(&As[(wm + 16 + l15) * 64 + 0], hi);
    v16bf a11 = loadFragA(&As[(wm + 16 + l15) * 64 + 32], hi);
    v16bf b00 = loadFragB(&Bt[(wn + l15) * 64 + 0], hi);
    v16bf b01 = loadFragB(&Bt[(wn + l15) * 64 + 32], hi);
    v16bf b10 = loadFragB(&Bt[(wn + 16 + l15) * 64 + 0], hi);
    v16bf b11 = loadFragB(&Bt[(wn + 16 + l15) * 64 + 32], hi);
    acc[0][0].v = wmma_bf16(a00, b00, acc[0][0].v);
    acc[0][0].v = wmma_bf16(a01, b01, acc[0][0].v);
    acc[0][1].v = wmma_bf16(a00, b10, acc[0][1].v);
    acc[0][1].v = wmma_bf16(a01, b11, acc[0][1].v);
    acc[1][0].v = wmma_bf16(a10, b00, acc[1][0].v);
    acc[1][0].v = wmma_bf16(a11, b01, acc[1][0].v);
    acc[1][1].v = wmma_bf16(a10, b10, acc[1][1].v);
    acc[1][1].v = wmma_bf16(a11, b11, acc[1][1].v);
    __syncthreads();
  }

#pragma unroll
  for (int mi = 0; mi < 2; ++mi)
#pragma unroll
    for (int ni = 0; ni < 2; ++ni)
#pragma unroll
      for (int r = 0; r < 8; ++r) {
        int m = bM + wm + mi * 16 + (hi ? 8 : 0) + r;
        int n = bN + wn + ni * 16 + l15;
        out[(size_t)m * EMB + n] = acc[mi][ni].f[r] + bout[n];
      }
}

// ---------------------------------------------------------------- launcher
extern "C" void kernel_launch(void* const* d_in, const int* in_sizes, int n_in,
                              void* d_out, int out_size, void* d_ws, size_t ws_size,
                              hipStream_t stream) {
  const float* x    = (const float*)d_in[0];
  const float* Wqkv = (const float*)d_in[1];
  const float* bqkv = (const float*)d_in[2];
  const float* Wout = (const float*)d_in[3];
  const float* bout = (const float*)d_in[4];
  float* out = (float*)d_out;

  char* ws = (char*)d_ws;
  unsigned short* wqkv_bf = (unsigned short*)(ws);                          // 6 MB
  unsigned short* wout_bf = (unsigned short*)(ws + 6u  * 1024 * 1024);      // 2 MB
  unsigned short* qws     = (unsigned short*)(ws + 8u  * 1024 * 1024);      // 8 MB
  unsigned short* kws     = (unsigned short*)(ws + 16u * 1024 * 1024);      // 8 MB
  unsigned short* vws     = (unsigned short*)(ws + 24u * 1024 * 1024);      // 8 MB
  unsigned short* ctx     = (unsigned short*)(ws + 32u * 1024 * 1024);      // 8 MB

  convert_f32_bf16<<<512, 256, 0, stream>>>(Wqkv, wqkv_bf, EMB * 3 * EMB);
  convert_f32_bf16<<<256, 256, 0, stream>>>(Wout, wout_bf, EMB * EMB);
  qkv_gemm<<<dim3(3 * EMB / 64, S_LEN / 128), 256, 0, stream>>>(
      x, wqkv_bf, bqkv, qws, kws, vws);
  flash_attn<<<dim3(S_LEN / 64, NH), 128, 0, stream>>>(qws, kws, vws, ctx);
  out_gemm<<<dim3(EMB / 64, S_LEN / 128), 256, 0, stream>>>(ctx, wout_bf, bout, out);
}
